// EncoderLSTM_89197880803708
// MI455X (gfx1250) — matrix operand order repaired and checked by reading, not compiled, not run on hardware
//
#include <hip/hip_runtime.h>
#include <hip/hip_bf16.h>

typedef unsigned short bf16_t;
typedef __attribute__((ext_vector_type(16))) __bf16 v16bf;
typedef __attribute__((ext_vector_type(8)))  __bf16 v8bf;
typedef __attribute__((ext_vector_type(8)))  float  v8f;
typedef __attribute__((ext_vector_type(4)))  unsigned int v4u;
typedef __attribute__((ext_vector_type(8)))  int v8i;
typedef __attribute__((ext_vector_type(4)))  int v4i;

#define B_  64
#define T_  1024
#define I_  256
#define H_  512
#define G_  2048   // 4*H
#define CHUNK_K 32
#define CHUNK_BYTES (G_ * CHUNK_K * 2)   // 128 KiB per weight chunk

static __device__ __forceinline__ bf16_t f32_to_bf16_bits(float f) {
  union { float f; unsigned int u; } v; v.f = f;
  unsigned int r = v.u + 0x7FFFu + ((v.u >> 16) & 1u);   // round-to-nearest-even
  return (bf16_t)(r >> 16);
}

static __device__ __forceinline__ float sigmoidf_(float x) {
  return 1.0f / (1.0f + __expf(-x));
}

// LDS byte offset of a generic shared pointer: per the CDNA5 aperture rule,
// addr[31:0] of a generic LDS address IS the wave-relative LDS offset.
static __device__ __forceinline__ unsigned lds_offset_of(const void* p) {
  return (unsigned)(unsigned long long)(uintptr_t)p;
}

// Issue one TDM load: 2-D tile (tile_dim0 = CHUNK_K along k, tile_dim1 = G_
// along gate columns) from row-major (g, k) bf16 weights into LDS, packed
// contiguously as LDS[(g*CHUNK_K + k')] bf16. This toolchain's builtin takes
// 6 args: (uint32x4 g0, int32x8 g1, int32x4 g2, int32x4 g3, int32x8 g4, cpol).
static __device__ __forceinline__ void tdm_load_chunk(const bf16_t* gsrc,
                                                      unsigned lds_off,
                                                      int kt) {
  unsigned long long ga = (unsigned long long)(uintptr_t)gsrc;
  v4u g0;
  g0[0] = 1u;                                         // count=1, user mode
  g0[1] = lds_off;                                    // lds_addr (bytes)
  g0[2] = (unsigned)(ga & 0xFFFFFFFFull);             // global_addr[31:0]
  g0[3] = (unsigned)((ga >> 32) & 0x01FFFFFFull)      // global_addr[56:32]
        | 0x80000000u;                                // type=2 ("image")
  v8i g1;
  g1[0] = 0x00010000;                                 // wg_mask=0, data_size=2B
  g1[1] = (int)((unsigned)(kt & 0xFFFF) << 16);       // tensor_dim0[15:0]
  g1[2] = (int)(((unsigned)(kt >> 16) & 0xFFFFu)      // tensor_dim0[31:16]
        | ((unsigned)(G_ & 0xFFFF) << 16));           // tensor_dim1[15:0]
  g1[3] = (int)((unsigned)CHUNK_K << 16);             // tensor_dim1 hi=0 | tile_dim0
  g1[4] = (int)G_;                                    // tile_dim1=2048, tile_dim2=0
  g1[5] = kt;                                         // tensor_dim0_stride[31:0]
  g1[6] = 0;                                          // stride hi | dim1_stride lo
  g1[7] = 0;
  v4i z4; z4[0] = 0; z4[1] = 0; z4[2] = 0; z4[3] = 0; // groups 2/3: unused (2-D)
  v8i z8;
  #pragma unroll
  for (int i = 0; i < 8; ++i) z8[i] = 0;              // group 4: unused
  __builtin_amdgcn_tensor_load_to_lds(g0, g1, z4, z4, z8, 0);
}

// One-shot: fp32 [W_ih | W_hh] -> bf16 row-major (g, k) + bias sum.
__global__ void pack_weights_kernel(const float* __restrict__ Wih,
                                    const float* __restrict__ Whh,
                                    const float* __restrict__ bih,
                                    const float* __restrict__ bhh,
                                    bf16_t* __restrict__ Wp,
                                    float* __restrict__ bsum,
                                    int in_dim) {
  const int KT = in_dim + H_;
  const int total = G_ * KT;
  int idx = blockIdx.x * blockDim.x + threadIdx.x;
  if (idx < total) {
    int g = idx / KT;
    int k = idx - g * KT;
    float w = (k < in_dim) ? Wih[g * in_dim + k] : Whh[g * H_ + (k - in_dim)];
    Wp[idx] = f32_to_bf16_bits(w);
  }
  if (idx < G_) bsum[idx] = bih[idx] + bhh[idx];
}

// Persistent fused LSTM layer. One workgroup per 16 batch rows (independent
// recurrence chains). 16 waves; wave w owns hidden slice [32w, 32w+32) so all
// four gates for that slice stay in its registers. Weight matrix streamed
// through LDS by the Tensor Data Mover, double-buffered in 128 KiB chunks.
template<int IN_DIM, bool FIRST>
__global__ __launch_bounds__(512, 1)
void lstm_layer_kernel(const float*  __restrict__ x_f32,  // FIRST : x (B,I,T) fp32
                       const bf16_t* __restrict__ x_bf,   // !FIRST: ys0 (B,T,H) bf16
                       const bf16_t* __restrict__ Wp,     // (2048, KT) bf16
                       const float*  __restrict__ bsum,   // (2048,) b_ih+b_hh
                       bf16_t* __restrict__ out_bf,       // FIRST : ys (B,T,H) bf16
                       float*  __restrict__ out_f32,      // !FIRST: ys (B,T,H) fp32
                       float*  __restrict__ hT,           // (B,H)
                       float*  __restrict__ cT) {         // (B,H)
  constexpr int KT    = IN_DIM + H_;
  constexpr int AROW  = KT + 8;                // padded row stride (halves)
  constexpr int A_AL  = ((16 * AROW * 2) + 255) & ~255;
  constexpr int NC    = KT / CHUNK_K;          // k-chunks per step

  extern __shared__ char smem[];
  bf16_t* lds_a = (bf16_t*)smem;               // A operand: [x_t | h_t] bf16
  bf16_t* wbuf0 = (bf16_t*)(smem + A_AL);      // weight chunk double buffer
  bf16_t* wbuf1 = (bf16_t*)(smem + A_AL + CHUNK_BYTES);
  const unsigned wb_off0 = lds_offset_of(wbuf0);
  const unsigned wb_off1 = lds_offset_of(wbuf1);

  const int tid  = threadIdx.x;
  const int lane = tid & 31;
  const int wave = tid >> 5;
  const int m0   = blockIdx.x * 16;            // batch tile base
  const int j0   = wave * 32;                  // this wave's hidden slice
  const int nlo  = lane & 15;                  // N column / A row (M)
  const int half = lane >> 4;                  // lane half -> K sub-range

  // h_0 = 0: zero A tile, then stage x_0
  for (int i = tid; i < 16 * AROW; i += 512) lds_a[i] = 0;
  __syncthreads();
  {
    const int m = tid >> 5;
    if (FIRST) {
      const int i0 = (tid & 31) * (IN_DIM / 32);
      #pragma unroll
      for (int ii = 0; ii < IN_DIM / 32; ++ii) {
        float v = x_f32[((size_t)(m0 + m) * I_ + (i0 + ii)) * (size_t)T_ + 0];
        lds_a[m * AROW + i0 + ii] = f32_to_bf16_bits(v);
      }
    } else {
      const int c0 = (tid & 31) * 16;
      const uint4* src = (const uint4*)(x_bf + ((size_t)(m0 + m) * T_ + 0) * H_ + c0);
      uint4 a = src[0], b = src[1];
      *(uint4*)&lds_a[m * AROW + c0]     = a;
      *(uint4*)&lds_a[m * AROW + c0 + 8] = b;
    }
  }

  float bias[4][2];
  #pragma unroll
  for (int gate = 0; gate < 4; ++gate)
    #pragma unroll
    for (int tt = 0; tt < 2; ++tt)
      bias[gate][tt] = bsum[gate * H_ + j0 + tt * 16 + nlo];

  float cst[16];
  #pragma unroll
  for (int i = 0; i < 16; ++i) cst[i] = 0.0f;

  for (int t = 0; t < T_; ++t) {
    // Prologue: DMA chunk 0 for this step (previous step's readers of wbuf0
    // are past a barrier already).
    if (wave == 0) tdm_load_chunk(Wp, wb_off0, KT);

    v8f acc[8];                                 // q = gate*2 + tt
    #pragma unroll
    for (int q = 0; q < 8; ++q)
      #pragma unroll
      for (int e = 0; e < 8; ++e) acc[q][e] = 0.0f;

    #pragma unroll 2
    for (int c = 0; c < NC; ++c) {
      const int kc = c * CHUNK_K;
      if (wave == 0) {
        if (c + 1 < NC) {
          tdm_load_chunk(Wp + (size_t)(c + 1) * CHUNK_K,
                         ((c + 1) & 1) ? wb_off1 : wb_off0, KT);
          __builtin_amdgcn_s_wait_tensorcnt(1);   // chunk c complete (in-order)
        } else {
          __builtin_amdgcn_s_wait_tensorcnt(0);
        }
      }
      __syncthreads();                            // chunk c (and, for c==0, x_t/h_t) visible

      // A fragment (16-bit A 16x32): lane holds row M=nlo;
      // elems 0..7 = K kc+8*half.., elems 8..15 = K kc+16+8*half..
      union { v16bf v; v8bf h[2]; } afrag;
      afrag.h[0] = *(const v8bf*)&lds_a[nlo * AROW + kc + half * 8];
      afrag.h[1] = *(const v8bf*)&lds_a[nlo * AROW + kc + 16 + half * 8];

      // All 8 B fragments into distinct registers, then 8 WMMAs.
      const bf16_t* wb = (c & 1) ? wbuf1 : wbuf0;
      union { v16bf v; v8bf h[2]; } bfr[8];
      #pragma unroll
      for (int q = 0; q < 8; ++q) {
        const int gate = q >> 1, tt = q & 1;
        const int g = gate * H_ + j0 + tt * 16 + nlo;
        const bf16_t* p = wb + (size_t)g * CHUNK_K + half * 8;
        bfr[q].h[0] = *(const v8bf*)(p);
        bfr[q].h[1] = *(const v8bf*)(p + 16);
      }
      #pragma unroll
      for (int q = 0; q < 8; ++q)
        acc[q] = __builtin_amdgcn_wmma_f32_16x16x32_bf16(
            false, afrag.v, false, bfr[q].v, (short)0, acc[q], false, false);

      __syncthreads();                            // readers done before buffer reuse
    }

    // Gate nonlinearities + state update (D layout: lane -> N=nlo,
    // vgpr e -> M = e + 8*half), all wave-local.
    #pragma unroll
    for (int tt = 0; tt < 2; ++tt) {
      const int j = j0 + tt * 16 + nlo;
      #pragma unroll
      for (int e = 0; e < 8; ++e) {
        const int m = e + (half << 3);
        float iv = sigmoidf_(acc[0 * 2 + tt][e] + bias[0][tt]);
        float fv = sigmoidf_(acc[1 * 2 + tt][e] + bias[1][tt]);
        float gv = tanhf    (acc[2 * 2 + tt][e] + bias[2][tt]);
        float ov = sigmoidf_(acc[3 * 2 + tt][e] + bias[3][tt]);
        float cv = fv * cst[tt * 8 + e] + iv * gv;
        cst[tt * 8 + e] = cv;
        float hv = ov * tanhf(cv);
        bf16_t hb = f32_to_bf16_bits(hv);
        lds_a[m * AROW + IN_DIM + j] = hb;       // h_{t+1} -> next step's A operand
        if (FIRST) {
          out_bf[((size_t)(m0 + m) * T_ + t) * H_ + j] = hb;
        } else {
          out_f32[((size_t)(m0 + m) * T_ + t) * H_ + j] = hv;
        }
        if (t == T_ - 1) {
          hT[(size_t)(m0 + m) * H_ + j] = hv;
          cT[(size_t)(m0 + m) * H_ + j] = cv;
        }
      }
    }

    // Stage x_{t+1} (disjoint LDS region from h writes above). Visibility to
    // other waves is guaranteed by the c==0 barrier of the next iteration.
    if (t + 1 < T_) {
      const int m = tid >> 5;
      if (FIRST) {
        const int i0 = (tid & 31) * (IN_DIM / 32);
        #pragma unroll
        for (int ii = 0; ii < IN_DIM / 32; ++ii) {
          const size_t base = ((size_t)(m0 + m) * I_ + (i0 + ii)) * (size_t)T_;
          float v = x_f32[base + (t + 1)];
          if (t + 2 < T_) __builtin_prefetch(&x_f32[base + (t + 2)], 0, 1);
          lds_a[m * AROW + i0 + ii] = f32_to_bf16_bits(v);
        }
      } else {
        const int c0 = (tid & 31) * 16;
        const uint4* src = (const uint4*)(x_bf + ((size_t)(m0 + m) * T_ + (t + 1)) * H_ + c0);
        uint4 a = src[0], b = src[1];
        *(uint4*)&lds_a[m * AROW + c0]     = a;
        *(uint4*)&lds_a[m * AROW + c0 + 8] = b;
      }
    }
  }
}

extern "C" void kernel_launch(void* const* d_in, const int* in_sizes, int n_in,
                              void* d_out, int out_size, void* d_ws, size_t ws_size,
                              hipStream_t stream) {
  (void)in_sizes; (void)n_in; (void)out_size; (void)ws_size;
  const float* x    = (const float*)d_in[0];
  const float* Wih0 = (const float*)d_in[1];
  const float* Whh0 = (const float*)d_in[2];
  const float* bih0 = (const float*)d_in[3];
  const float* bhh0 = (const float*)d_in[4];
  const float* Wih1 = (const float*)d_in[5];
  const float* Whh1 = (const float*)d_in[6];
  const float* bih1 = (const float*)d_in[7];
  const float* bhh1 = (const float*)d_in[8];

  char* ws = (char*)d_ws;
  size_t off = 0;
  bf16_t* Wp0 = (bf16_t*)(ws + off); off += (size_t)G_ * (I_ + H_) * sizeof(bf16_t); // 3 MiB
  bf16_t* Wp1 = (bf16_t*)(ws + off); off += (size_t)G_ * (H_ + H_) * sizeof(bf16_t); // 4 MiB
  float*  bs0 = (float*)(ws + off);  off += (size_t)G_ * sizeof(float);
  float*  bs1 = (float*)(ws + off);  off += (size_t)G_ * sizeof(float);
  off = (off + 255) & ~(size_t)255;
  bf16_t* ys0 = (bf16_t*)(ws + off); off += (size_t)B_ * T_ * H_ * sizeof(bf16_t);   // 64 MiB

  float* out  = (float*)d_out;
  float* hid  = out + (size_t)B_ * T_ * H_;
  float* cell = hid + 2ull * B_ * H_;

  {
    int total0 = G_ * (I_ + H_);
    pack_weights_kernel<<<(total0 + 255) / 256, 256, 0, stream>>>(
        Wih0, Whh0, bih0, bhh0, Wp0, bs0, I_);
    int total1 = G_ * (H_ + H_);
    pack_weights_kernel<<<(total1 + 255) / 256, 256, 0, stream>>>(
        Wih1, Whh1, bih1, bhh1, Wp1, bs1, H_);
  }

  // Dynamic LDS: A tile + 2 x 128 KiB weight chunk buffers (fits in CDNA5's
  // 320 KiB/WGP; exceeds the old 64 KiB static limit, hence dynamic).
  constexpr int SM0 = ((16 * (I_ + H_ + 8) * 2 + 255) & ~255) + 2 * CHUNK_BYTES; // ~287 KiB
  constexpr int SM1 = ((16 * (H_ + H_ + 8) * 2 + 255) & ~255) + 2 * CHUNK_BYTES; // ~295 KiB

  lstm_layer_kernel<I_, true><<<B_ / 16, 512, SM0, stream>>>(
      x, nullptr, Wp0, bs0, ys0, nullptr, hid, cell);

  lstm_layer_kernel<H_, false><<<B_ / 16, 512, SM1, stream>>>(
      nullptr, ys0, Wp1, bs1, nullptr, out,
      hid + (size_t)B_ * H_, cell + (size_t)B_ * H_);
}
